// GraphConvolution_60533269070024
// MI455X (gfx1250) — matrix operand order repaired and checked by reading, not compiled, not run on hardware
//
#include <hip/hip_runtime.h>

// ---------------------------------------------------------------------------
// Fused GraphConvolution for MI455X (gfx1250, wave32)
//   out = input @ W_self + (adj @ input) @ W_agg
// N = 16384, F = 128.
//  - adj (1 GiB, HBM-bound stream) tiles DMA'd by the Tensor Data Mover
//    (tensor_load_to_lds, TENSORcnt), double-buffered in LDS as f32.
//  - input tiles (L2-resident) staged K-transposed as f16 via VGPR convert.
//  - epilogue input tiles via GLOBAL_LOAD_ASYNC_TO_LDS_B128 (ASYNCcnt).
//  - all MACs on v_wmma_f32_16x16x32_f16.
// ---------------------------------------------------------------------------

#define NN 16384
#define F  128
#define BM 64      // rows of out per block
#define BK 64      // K-chunk per main-loop iteration (2 WMMA k-steps)

// LDS layout (byte offsets; dynamic LDS starts at LDS address 0)
#define ABUF_OFF   0        // f32 adj/input tile, double buffered: 2 x 64x64x4 = 32 KB
#define BT_OFF     32768    // f16 input^T tile / Sbuf: 16 KB
#define WT_OFF     49152    // f16 Wself^T + Wagg^T: 64 KB
#define LDS_BYTES  114688

typedef _Float16     v16h  __attribute__((ext_vector_type(16)));
typedef float        v8f   __attribute__((ext_vector_type(8)));
typedef unsigned int u32x4 __attribute__((ext_vector_type(4)));
typedef int          i32x4 __attribute__((ext_vector_type(4)));
typedef int          i32x8 __attribute__((ext_vector_type(8)));

#if defined(__has_builtin)
#if __has_builtin(__builtin_amdgcn_tensor_load_to_lds)
#define HAVE_TDM 1
#endif
#endif

union Frag16 {
    v16h  h;
    uint4 q[2];
};
union Frag32 {
    float f[16];
    uint4 q[4];
};

// 16 f16 elements as two 16B LDS reads (ds_load_b128).
__device__ __forceinline__ v16h ld_frag16(const _Float16* rowbase, int q0, int q1) {
    Frag16 t;
    const uint4* p = (const uint4*)rowbase;
    t.q[0] = p[q0];
    t.q[1] = p[q1];
    return t.h;
}

// 16 f32 elements (two 8-float runs at e0,e1) -> f16 fragment.
__device__ __forceinline__ v16h ld_frag32(const float* rowbase, int e0, int e1) {
    Frag32 t;
    const uint4* p = (const uint4*)rowbase;
    t.q[0] = p[(e0 >> 2) + 0];
    t.q[1] = p[(e0 >> 2) + 1];
    t.q[2] = p[(e1 >> 2) + 0];
    t.q[3] = p[(e1 >> 2) + 1];
    v16h a;
    #pragma unroll
    for (int e = 0; e < 16; ++e) a[e] = (_Float16)t.f[e];
    return a;
}

// Per-lane async DMA: 16 bytes global -> LDS, tracked by ASYNCcnt.
__device__ __forceinline__ void async_copy_b128(unsigned lds_off, const void* gaddr) {
    asm volatile("global_load_async_to_lds_b128 %0, %1, off"
                 :
                 : "v"(lds_off), "v"((unsigned long long)(size_t)gaddr)
                 : "memory");
}

// TDM: DMA a 2D f32 tile (tile_w x tile_h elems) from a row-major tensor
// (row stride = stride_elems) into LDS at lds_off. Tracked by TENSORcnt.
__device__ __forceinline__ void tdm_load_tile_f32(
    unsigned lds_off, const void* gtile, unsigned stride_elems,
    unsigned tile_w, unsigned tile_h)
{
#ifdef HAVE_TDM
    unsigned long long ga = (unsigned long long)(size_t)gtile;
    u32x4 g0;
    g0.x = 1u;                                      // count=1 (valid user D#)
    g0.y = lds_off;                                 // lds_addr (bytes)
    g0.z = (unsigned)ga;                            // global_addr[31:0]
    g0.w = (unsigned)(ga >> 32) | (2u << 30);       // global_addr[56:32] | type=2
    i32x8 g1;
    g1[0] = (int)(2u << 16);                        // data_size = 4B; no mask/pad
    g1[1] = (int)(stride_elems << 16);              // tensor_dim0[15:0] (= stride)
    g1[2] = (int)((stride_elems >> 16) | ((unsigned)NN << 16)); // dim0 hi | dim1 lo
    g1[3] = (int)(((unsigned)NN >> 16) | (tile_w << 16));       // dim1 hi | tile_dim0
    g1[4] = (int)tile_h;                            // tile_dim1 (tile_dim2 = 0)
    g1[5] = (int)stride_elems;                      // tensor_dim0_stride[31:0]
    g1[6] = 0;                                      // stride[47:32] | dim1_stride lo
    g1[7] = 0;
    i32x4 z4 = {0, 0, 0, 0};
#if __clang_major__ >= 23
    i32x8 z8 = {0, 0, 0, 0, 0, 0, 0, 0};
    __builtin_amdgcn_tensor_load_to_lds(g0, g1, z4, z4, z8, 0);
#else
    __builtin_amdgcn_tensor_load_to_lds(g0, g1, z4, z4, 0);
#endif
#else
    (void)lds_off; (void)gtile; (void)stride_elems; (void)tile_w; (void)tile_h;
#endif
}

__device__ __forceinline__ void wait_tensor_le1() {
    asm volatile("s_wait_tensorcnt 0x1" ::: "memory");
}
__device__ __forceinline__ void wait_tensor_0() {
    asm volatile("s_wait_tensorcnt 0x0" ::: "memory");
}
__device__ __forceinline__ void wait_async_le4() {
    asm volatile("s_wait_asynccnt 0x4" ::: "memory");
}
__device__ __forceinline__ void wait_async_0() {
    asm volatile("s_wait_asynccnt 0x0" ::: "memory");
}

__global__ __launch_bounds__(256) void gcn_fused_wmma(
    const float* __restrict__ input,    // [N, F]
    const float* __restrict__ adj,      // [N, N]
    const float* __restrict__ weight,   // [2F, F]
    float* __restrict__ out)            // [N, F]
{
    extern __shared__ char smem[];
    float*    Abuf = (float*)(smem + ABUF_OFF);      // [2][64][64] f32
    _Float16* Bt   = (_Float16*)(smem + BT_OFF);     // [128][64] f16 / Sbuf [64][128]
    _Float16* Wt   = (_Float16*)(smem + WT_OFF);     // Wself^T then Wagg^T, [128][128] f16

    const int tid  = threadIdx.x;
    const int lane = tid & 31;
    const int wave = tid >> 5;
    const int g    = lane >> 4;        // half-wave group (ISA A/B layout)
    const int ln   = lane & 15;
    const int mrow = (wave & 3) * 16;  // wave's row tile inside block
    const int cgrp = (wave >> 2) * 64; // wave's column group
    const int row_base = blockIdx.x * BM;

    const float4* in4 = (const float4*)input;
    const float4* w4  = (const float4*)weight;

    // per-thread 16B-chunk coords for per-lane async fills (64 rows x 256 B)
    const int arow = tid >> 4;         // 0..15 (row; +16*i)
    const int ac16 = tid & 15;         // 0..15 (16B chunk within 256B row)

    // ---- stage both weight halves into LDS, transposed (Wt[n][k]), f16 ----
    for (int i = 0; i < 32; ++i) {
        int flat = tid + 256 * i;
        int kw   = flat >> 5;
        int c4   = flat & 31;
        float4 v = w4[kw * (F / 4) + c4];
        _Float16* dst = Wt + ((kw < F) ? 0 : (F * F));
        int k2 = (kw < F) ? kw : (kw - F);
        dst[(c4 * 4 + 0) * F + k2] = (_Float16)v.x;
        dst[(c4 * 4 + 1) * F + k2] = (_Float16)v.y;
        dst[(c4 * 4 + 2) * F + k2] = (_Float16)v.z;
        dst[(c4 * 4 + 3) * F + k2] = (_Float16)v.w;
    }

    // ---- prologue: DMA first adj tile into buffer 0 ----
#ifdef HAVE_TDM
    if (wave == 0) {
        tdm_load_tile_f32(ABUF_OFF, adj + (size_t)row_base * NN, NN, BK, BM);
    }
#else
    #pragma unroll
    for (int i = 0; i < 4; ++i) {
        int r = arow + 16 * i;
        async_copy_b128(ABUF_OFF + r * 256 + ac16 * 16,
                        (const char*)adj + ((size_t)(row_base + r) * NN) * 4 + ac16 * 16);
    }
#endif

    // ---- main loop: S(64x128) = adj[rows,:] @ input via WMMA ----
    v8f acc[4] = {};
    for (int k0 = 0; k0 < NN; k0 += BK) {
        const int p = (k0 / BK) & 1;
        __syncthreads();   // previous compute done: safe to refill Bt / buf p^1
        const bool has_next = (k0 + BK) < NN;
        if (has_next) {
#ifdef HAVE_TDM
            if (wave == 0) {
                tdm_load_tile_f32(ABUF_OFF + (p ^ 1) * 16384,
                                  adj + (size_t)row_base * NN + (k0 + BK),
                                  NN, BK, BM);
            }
#else
            #pragma unroll
            for (int i = 0; i < 4; ++i) {
                int r = arow + 16 * i;
                async_copy_b128(ABUF_OFF + (p ^ 1) * 16384 + r * 256 + ac16 * 16,
                                (const char*)adj +
                                    ((size_t)(row_base + r) * NN + (k0 + BK)) * 4 + ac16 * 16);
            }
#endif
        }
        // input tile (64 x 128 f32) -> K-transposed f16: Bt[n][k]
        {
            int kr = tid >> 5;
            int n4 = tid & 31;
            for (int i = 0; i < 8; ++i) {
                int k = kr + 8 * i;
                float4 v = in4[(size_t)(k0 + k) * (F / 4) + n4];
                Bt[(n4 * 4 + 0) * BK + k] = (_Float16)v.x;
                Bt[(n4 * 4 + 1) * BK + k] = (_Float16)v.y;
                Bt[(n4 * 4 + 2) * BK + k] = (_Float16)v.z;
                Bt[(n4 * 4 + 3) * BK + k] = (_Float16)v.w;
            }
        }
        // current tile complete (next stays in flight), then block-wide barrier
#ifdef HAVE_TDM
        if (wave == 0) {
            if (has_next) wait_tensor_le1();
            else          wait_tensor_0();
        }
#else
        if (has_next) wait_async_le4();
        else          wait_async_0();
#endif
        __syncthreads();

        const float* Acur = Abuf + p * 4096;   // f32 view of current tile
        for (int kk = 0; kk < BK; kk += 32) {
            int qb = kk >> 3;
            v16h a = ld_frag32(Acur + (mrow + ln) * BK, kk + 8 * g, kk + 16 + 8 * g);
            #pragma unroll
            for (int ct = 0; ct < 4; ++ct) {
                const _Float16* brow = Bt + (cgrp + ct * 16 + ln) * BK;
                v16h b = ld_frag16(brow, qb + g, qb + 2 + g);
                acc[ct] = __builtin_amdgcn_wmma_f32_16x16x32_f16(
                    false, a, false, b, (short)0, acc[ct], false, false);
            }
        }
    }

    // ---- spill S to LDS as f16 (A-operand source for epilogue) ----
    __syncthreads();
    _Float16* Sbuf = Bt;               // reuse: [64][128] f16
    #pragma unroll
    for (int ct = 0; ct < 4; ++ct) {
        int n = cgrp + ct * 16 + ln;
        #pragma unroll
        for (int r = 0; r < 8; ++r) {
            int m = mrow + 8 * g + r;  // C/D layout: VGPR r, half-wave g
            Sbuf[m * F + n] = (_Float16)acc[ct][r];
        }
    }

    // ---- epilogue: out = input_rows @ Wself + S @ Wagg  (K = 128) ----
    v8f oacc[4] = {};
    for (int kb = 0; kb < 2; ++kb) {
        __syncthreads();               // prior reads of Abuf done; DMA may overwrite
        #pragma unroll
        for (int i = 0; i < 4; ++i) {
            int r = arow + 16 * i;
            async_copy_b128(ABUF_OFF + r * 256 + ac16 * 16,
                            (const char*)input +
                                ((size_t)(row_base + r) * F + kb * BK) * 4 + ac16 * 16);
        }
        wait_async_0();
        __syncthreads();
        for (int kk = 0; kk < BK; kk += 32) {
            int kglob = kb * BK + kk;
            int qg = kglob >> 3;
            v16h ain = ld_frag32(Abuf + (mrow + ln) * BK, kk + 8 * g, kk + 16 + 8 * g);
            v16h as  = ld_frag16(Sbuf + (mrow + ln) * F,  qg + g, qg + 2 + g);
            #pragma unroll
            for (int ct = 0; ct < 4; ++ct) {
                int n = cgrp + ct * 16 + ln;
                v16h bs = ld_frag16(Wt + n * F,         qg + g, qg + 2 + g);
                oacc[ct] = __builtin_amdgcn_wmma_f32_16x16x32_f16(
                    false, ain, false, bs, (short)0, oacc[ct], false, false);
                v16h ba = ld_frag16(Wt + F * F + n * F, qg + g, qg + 2 + g);
                oacc[ct] = __builtin_amdgcn_wmma_f32_16x16x32_f16(
                    false, as, false, ba, (short)0, oacc[ct], false, false);
            }
        }
    }

    // ---- store out (f32), C/D layout -> global ----
    #pragma unroll
    for (int ct = 0; ct < 4; ++ct) {
        int n = cgrp + ct * 16 + ln;
        #pragma unroll
        for (int r = 0; r < 8; ++r) {
            int m = row_base + mrow + 8 * g + r;
            out[(size_t)m * F + n] = oacc[ct][r];
        }
    }
}

extern "C" void kernel_launch(void* const* d_in, const int* in_sizes, int n_in,
                              void* d_out, int out_size, void* d_ws, size_t ws_size,
                              hipStream_t stream) {
    const float* input  = (const float*)d_in[0];   // [N, F]
    const float* adj    = (const float*)d_in[1];   // [N, N]
    const float* weight = (const float*)d_in[2];   // [2F, F]
    float* out = (float*)d_out;                    // [N, F]

    static_assert(LDS_BYTES == 32768 + 16384 + 65536, "lds layout");

    hipFuncSetAttribute(reinterpret_cast<const void*>(gcn_fused_wmma),
                        hipFuncAttributeMaxDynamicSharedMemorySize, (int)LDS_BYTES);

    dim3 grid(NN / BM);   // 256 blocks, each owns 64 output rows
    dim3 block(256);      // 8 wave32 waves
    gcn_fused_wmma<<<grid, block, LDS_BYTES, stream>>>(input, adj, weight, out);
}
